// SNNCore_74010876445349
// MI455X (gfx1250) — matrix operand order repaired and checked by reading, not compiled, not run on hardware
//
#include <hip/hip_runtime.h>

// ---------------------------------------------------------------------------
// SNN predictive-coding forward for MI455X (gfx1250, wave32, WMMA bf16)
//   B=8 S=256 D=512 V=32000 L=4 T=4
// ---------------------------------------------------------------------------

#define B_ 8
#define S_ 256
#define D_ 512
#define V_ 32000
#define L_ 4
#define T_ 4
#define LD_ (L_ * D_)        // 2048
#define DECAY_ 0.95f
#define THR_ 1.0f
#define EPS_ 1e-5f

typedef __attribute__((ext_vector_type(16))) __bf16 v16bf;
typedef __attribute__((ext_vector_type(8)))  __bf16 v8bf;
typedef __attribute__((ext_vector_type(8)))  float  v8f;

union V16U { v16bf v; v8bf h[2]; };
union V8U  { v8bf  v; unsigned short s[8]; };

__device__ __forceinline__ unsigned short f2bf(float f) {
    unsigned int u = __float_as_uint(f);
    return (unsigned short)((u + 0x7FFFu + ((u >> 16) & 1u)) >> 16);
}

__device__ __forceinline__ v8f vzero() {
    v8f z = {0.f, 0.f, 0.f, 0.f, 0.f, 0.f, 0.f, 0.f};
    return z;
}

__device__ __forceinline__ v8f wmma_bf16(v16bf a, v16bf b, v8f c) {
    // v_wmma_f32_16x16x32_bf16  (neg_a, A, neg_b, B, c_mod, C, reuse_a, reuse_b)
    return __builtin_amdgcn_wmma_f32_16x16x32_bf16(false, a, false, b,
                                                   (short)0, c, false, false);
}

// A fragment (16x32 bf16, M x K), contiguous K order matching B:
// lane l<16: row l, K [k..k+15]
// lane l>=16: row l-16, K [k+16..k+31]
__device__ __forceinline__ v16bf ldA(const unsigned short* base, int strideUS,
                                     int kbase, int lane) {
    const int row = lane & 15, hi = lane >> 4;
    const unsigned short* p = base + row * strideUS + kbase + hi * 16;
    V16U u;
    u.h[0] = *(const v8bf*)p;
    u.h[1] = *(const v8bf*)(p + 8);
    return u.v;
}

// B fragment (32x16 bf16, K x N): lane l holds column n=l&15,
// 16 contiguous K halves starting at k + (l>>4)*16.  Since every GEMM here is
// X @ W^T, column n of B is row n of W -> contiguous 32B global load.
__device__ __forceinline__ v16bf ldB_glob(const unsigned short* W, int rowStrideUS,
                                          int nBase, int kBase, int lane) {
    const unsigned short* p = W + (size_t)(nBase + (lane & 15)) * rowStrideUS
                                + kBase + ((lane >> 4) << 4);
    return *(const v16bf*)p;
}

__device__ __forceinline__ v16bf ldB_lds(const unsigned short* b_sh, int strideUS,
                                         int nBase, int lane) {
    const unsigned short* p = b_sh + (size_t)(nBase + (lane & 15)) * strideUS
                                   + ((lane >> 4) << 4);
    V16U u;
    u.h[0] = *(const v8bf*)p;
    u.h[1] = *(const v8bf*)(p + 8);
    return u.v;
}

// ---------------------------------------------------------------------------
// Kernel 0: fp32 -> bf16 weight conversion + state zero-init
// ---------------------------------------------------------------------------
__global__ __launch_bounds__(256) void k_convert(
    const float* encW, const float* genW, const float* infW,
    unsigned short* encB, unsigned short* genB, unsigned short* infB,
    float* st, float* gm, float* im)
{
    const int i = blockIdx.x * 256 + threadIdx.x;
    if (i < D_ * D_)          encB[i] = f2bf(encW[i]);
    if (i < L_ * D_ * D_) {   genB[i] = f2bf(genW[i]);
                              infB[i] = f2bf(infW[i]); }
    if (i < L_ * B_ * D_) {   st[i] = 0.f; gm[i] = 0.f; im[i] = 0.f; }
}

// ---------------------------------------------------------------------------
// Kernel 1: encoder  x[m] = emb[ids] @ enc_W^T + enc_b   (m = s*8+b)
// one block = 16 rows x 512 cols, 8 waves each own 64 cols
// ---------------------------------------------------------------------------
__global__ __launch_bounds__(256) void k_encode(
    const int* ids, const float* emb, const unsigned short* encWb,
    const float* enc_b, float* x_g)
{
    __shared__ unsigned short a_sh[16 * 520];    // padded stride vs. LDS banks
    const int tid = threadIdx.x, lane = tid & 31, wave = tid >> 5;
    const int m0 = blockIdx.x * 16;

    for (int i = tid; i < 16 * D_; i += 256) {
        const int r = i >> 9, c = i & (D_ - 1);
        const int m = m0 + r, b = m & 7, s = m >> 3;
        const int tok = ids[b * S_ + s];
        a_sh[r * 520 + c] = f2bf(emb[(size_t)tok * D_ + c]);
    }
    __syncthreads();

    v8f acc[4];
    for (int i = 0; i < 4; ++i) acc[i] = vzero();
    const int n0 = wave * 64;
    for (int kk = 0; kk < D_; kk += 32) {
        const v16bf a = ldA(a_sh, 520, kk, lane);
        for (int sub = 0; sub < 4; ++sub) {
            const v16bf b = ldB_glob(encWb, D_, n0 + sub * 16, kk, lane);
            acc[sub] = wmma_bf16(a, b, acc[sub]);
        }
    }
    const int hi = lane >> 4;
    for (int sub = 0; sub < 4; ++sub) {
        const int n = n0 + sub * 16 + (lane & 15);
        const float bias = enc_b[n];
        for (int r = 0; r < 8; ++r) {
            const int m = m0 + hi * 8 + r;
            x_g[(size_t)m * D_ + n] = acc[sub][r] + bias;
        }
    }
}

// ---------------------------------------------------------------------------
// Kernel 2: persistent recurrent scan (single workgroup, 32 waves, 1 WGP)
// The scan over T*S = 1024 steps is strictly sequential; LIF / error updates
// are fused into each GEMM's epilogue (fixed element->thread mapping).
// ---------------------------------------------------------------------------
__device__ __forceinline__ void ln_row(const float* src, const float* g,
                                       const float* b, unsigned short* dst,
                                       int lane)
{
    float v[16];
    float sum = 0.f, sq = 0.f;
    for (int i = 0; i < 16; ++i) {
        const float x = src[lane + i * 32];
        v[i] = x; sum += x; sq += x * x;
    }
    for (int off = 16; off > 0; off >>= 1) {
        sum += __shfl_xor(sum, off);
        sq  += __shfl_xor(sq,  off);
    }
    const float mu   = sum * (1.f / 512.f);
    const float var  = sq * (1.f / 512.f) - mu * mu;
    const float rstd = rsqrtf(var + EPS_);
    for (int i = 0; i < 16; ++i) {
        const int c = lane + i * 32;
        dst[c] = f2bf((v[i] - mu) * rstd * g[c] + b[c]);
    }
}

__global__ __launch_bounds__(1024) void k_recurrent(
    const float* x_g,
    const unsigned short* genWb, const float* gen_b,
    const unsigned short* infWb, const float* inf_b,
    const float* ns_g, const float* ns_b,
    const float* ne_g, const float* ne_b,
    const float* err_scale,
    float* states, float* gmems, float* imems,
    unsigned short* finals)
{
    __shared__ unsigned short a_sh[16 * 520];   // LN output (rows 8..15 = 0 pad)
    __shared__ float bu0[B_ * D_];
    __shared__ float bu1[B_ * D_];
    const int tid = threadIdx.x, lane = tid & 31, wave = tid >> 5;

    for (int i = tid; i < 8 * 520; i += 1024) a_sh[8 * 520 + i] = 0;
    __syncthreads();

    float* buIn  = bu0;
    float* buOut = bu1;

    for (int t = 0; t < T_; ++t) {
        for (int s = 0; s < S_; ++s) {
            for (int i = tid; i < B_ * D_; i += 1024)
                buIn[i] = x_g[s * B_ * D_ + i];
            __syncthreads();

            for (int j = 0; j < L_; ++j) {
                // ---- Phase 1: ns = LN(state_j) -> a_sh (bf16) ----
                if (wave < 8)
                    ln_row(states + (j * B_ + wave) * D_,
                           ns_g + j * D_, ns_b + j * D_,
                           a_sh + wave * 520, lane);
                __syncthreads();

                // ---- Phase 2: pred_in = ns @ genW^T; LIF(gen); pe ----
                {
                    v8f acc = vzero();
                    const unsigned short* Wj = genWb + (size_t)j * D_ * D_;
                    for (int kk = 0; kk < D_; kk += 32) {
                        const v16bf a = ldA(a_sh, 520, kk, lane);
                        const v16bf b = ldB_glob(Wj, D_, wave * 16, kk, lane);
                        acc = wmma_bf16(a, b, acc);
                    }
                    if (lane < 16) {            // rows 8..15 are zero padding
                        const int n = wave * 16 + lane;
                        const float bias = gen_b[j * D_ + n];
                        const float esc  = err_scale[j];
                        for (int r = 0; r < 8; ++r) {
                            const int idx = (j * B_ + r) * D_ + n;
                            const float gm   = gmems[idx] * DECAY_ + acc[r] + bias;
                            const float pred = gm > THR_ ? 1.f : 0.f;
                            gmems[idx] = gm * (1.f - pred);
                            buOut[r * D_ + n] = (buIn[r * D_ + n] - pred) * esc;
                        }
                    }
                }
                __syncthreads();

                // ---- Phase 3: ne = LN(pe) -> a_sh (bf16) ----
                if (wave < 8)
                    ln_row(buOut + wave * D_,
                           ne_g + j * D_, ne_b + j * D_,
                           a_sh + wave * 520, lane);
                __syncthreads();

                // ---- Phase 4: inf_in = ne @ infW^T; LIF(inf); state ----
                {
                    v8f acc = vzero();
                    const unsigned short* Wj = infWb + (size_t)j * D_ * D_;
                    for (int kk = 0; kk < D_; kk += 32) {
                        const v16bf a = ldA(a_sh, 520, kk, lane);
                        const v16bf b = ldB_glob(Wj, D_, wave * 16, kk, lane);
                        acc = wmma_bf16(a, b, acc);
                    }
                    if (lane < 16) {
                        const int n = wave * 16 + lane;
                        const float bias = inf_b[j * D_ + n];
                        for (int r = 0; r < 8; ++r) {
                            const int idx = (j * B_ + r) * D_ + n;
                            const float im = imems[idx] * DECAY_ + acc[r] + bias;
                            const float su = im > THR_ ? 1.f : 0.f;
                            imems[idx] = im * (1.f - su);
                            const float stn = states[idx] * 0.9f + su * 0.1f;
                            states[idx] = stn;
                            if (t == T_ - 1)
                                finals[(size_t)(s * B_ + r) * LD_ + j * D_ + n]
                                    = f2bf(stn);
                        }
                    }
                }
                __syncthreads();

                float* tmp = buIn; buIn = buOut; buOut = tmp;  // bu = pe
            }
        }
    }
}

// ---------------------------------------------------------------------------
// Kernel 3: logits = finals @ out_W^T + out_b   [2048 x 2048] x [2048 x 32000]
// block tile 128x128, 8 waves (2x4) of 64x32, K-step 32, bf16 WMMA.
// out_W converted fp32->bf16 on the fly into LDS.
// ---------------------------------------------------------------------------
__global__ __launch_bounds__(256) void k_logits(
    const unsigned short* finals, const float* outW, const float* out_b,
    float* out)
{
    __shared__ unsigned short a_sh[128 * 40];
    __shared__ unsigned short b_sh[128 * 40];
    const int tid = threadIdx.x, lane = tid & 31, wave = tid >> 5;
    const int n0 = blockIdx.x * 128;
    const int m0 = blockIdx.y * 128;
    const int wm = wave >> 2, wn = wave & 3;

    v8f acc[4][2];
    for (int i = 0; i < 4; ++i)
        for (int j = 0; j < 2; ++j) acc[i][j] = vzero();

    const int rowA = tid >> 1;           // 0..127
    const int kcA  = (tid & 1) * 16;

    for (int kt = 0; kt < LD_; kt += 32) {
        {   // A tile (already bf16)
            const unsigned short* src =
                finals + (size_t)(m0 + rowA) * LD_ + kt + kcA;
            V16U u; u.v = *(const v16bf*)src;
            v8bf* d = (v8bf*)(a_sh + rowA * 40 + kcA);
            d[0] = u.h[0]; d[1] = u.h[1];
        }
        {   // B tile: row n of out_W, fp32 -> bf16
            const float* src = outW + (size_t)(n0 + rowA) * LD_ + kt + kcA;
            V8U lo, hi2;
            for (int i = 0; i < 8; ++i) {
                lo.s[i]  = f2bf(src[i]);
                hi2.s[i] = f2bf(src[i + 8]);
            }
            v8bf* d = (v8bf*)(b_sh + rowA * 40 + kcA);
            d[0] = lo.v; d[1] = hi2.v;
            if (kt + 32 < LD_) __builtin_prefetch(src + 32, 0, 1);
        }
        __syncthreads();

        v16bf bfr[2];
        for (int sub = 0; sub < 2; ++sub)
            bfr[sub] = ldB_lds(b_sh, 40, wn * 32 + sub * 16, lane);
        for (int ms = 0; ms < 4; ++ms) {
            const v16bf a = ldA(a_sh + (wm * 64 + ms * 16) * 40, 40, 0, lane);
            for (int sub = 0; sub < 2; ++sub)
                acc[ms][sub] = wmma_bf16(a, bfr[sub], acc[ms][sub]);
        }
        __syncthreads();
    }

    const int hi = lane >> 4;
    for (int ms = 0; ms < 4; ++ms)
        for (int sub = 0; sub < 2; ++sub) {
            const int n = n0 + wn * 32 + sub * 16 + (lane & 15);
            const float bias = out_b[n];
            for (int r = 0; r < 8; ++r) {
                const int m = m0 + wm * 64 + ms * 16 + hi * 8 + r;
                const int b = m & 7, sp = m >> 3;   // m = s*8 + b
                out[((size_t)b * S_ + sp) * (size_t)V_ + n]
                    = acc[ms][sub][r] + bias;
            }
        }
}

// ---------------------------------------------------------------------------
// Host launcher.  Workspace layout (~17 MB):
//   encW bf16 | genW bf16 | infW bf16 | x fp32 | states | gmems | imems | finals bf16
// ---------------------------------------------------------------------------
static constexpr size_t OFF_ENCB = 0;
static constexpr size_t OFF_GENB = OFF_ENCB + (size_t)D_ * D_ * 2;
static constexpr size_t OFF_INFB = OFF_GENB + (size_t)L_ * D_ * D_ * 2;
static constexpr size_t OFF_X    = OFF_INFB + (size_t)L_ * D_ * D_ * 2;
static constexpr size_t OFF_ST   = OFF_X    + (size_t)S_ * B_ * D_ * 4;
static constexpr size_t OFF_GM   = OFF_ST   + (size_t)L_ * B_ * D_ * 4;
static constexpr size_t OFF_IM   = OFF_GM   + (size_t)L_ * B_ * D_ * 4;
static constexpr size_t OFF_FIN  = OFF_IM   + (size_t)L_ * B_ * D_ * 4;

extern "C" void kernel_launch(void* const* d_in, const int* in_sizes, int n_in,
                              void* d_out, int out_size, void* d_ws,
                              size_t ws_size, hipStream_t stream)
{
    (void)in_sizes; (void)n_in; (void)out_size; (void)ws_size;

    const int*   ids   = (const int*)  d_in[0];
    const float* emb   = (const float*)d_in[1];
    const float* encW  = (const float*)d_in[2];
    const float* enc_b = (const float*)d_in[3];
    const float* genW  = (const float*)d_in[4];
    const float* gen_b = (const float*)d_in[5];
    const float* infW  = (const float*)d_in[6];
    const float* inf_b = (const float*)d_in[7];
    const float* ns_g  = (const float*)d_in[8];
    const float* ns_b  = (const float*)d_in[9];
    const float* ne_g  = (const float*)d_in[10];
    const float* ne_b  = (const float*)d_in[11];
    const float* esc   = (const float*)d_in[12];
    const float* outW  = (const float*)d_in[13];
    const float* out_b = (const float*)d_in[14];
    float* out = (float*)d_out;

    char* ws = (char*)d_ws;
    unsigned short* encB   = (unsigned short*)(ws + OFF_ENCB);
    unsigned short* genB   = (unsigned short*)(ws + OFF_GENB);
    unsigned short* infB   = (unsigned short*)(ws + OFF_INFB);
    float*          x_g    = (float*)(ws + OFF_X);
    float*          st_g   = (float*)(ws + OFF_ST);
    float*          gm_g   = (float*)(ws + OFF_GM);
    float*          im_g   = (float*)(ws + OFF_IM);
    unsigned short* fin_g  = (unsigned short*)(ws + OFF_FIN);

    k_convert<<<(L_ * D_ * D_ + 255) / 256, 256, 0, stream>>>(
        encW, genW, infW, encB, genB, infB, st_g, gm_g, im_g);

    k_encode<<<(B_ * S_) / 16, 256, 0, stream>>>(ids, emb, encB, enc_b, x_g);

    k_recurrent<<<1, 1024, 0, stream>>>(
        x_g, genB, gen_b, infB, inf_b,
        ns_g, ns_b, ne_g, ne_b, esc,
        st_g, gm_g, im_g, fin_g);

    k_logits<<<dim3(V_ / 128, (B_ * S_) / 128), 256, 0, stream>>>(
        fin_g, outW, out_b, out);
}
